// GCM_9414568312860
// MI455X (gfx1250) — matrix operand order repaired and checked by reading, not compiled, not run on hardware
//
#include <hip/hip_runtime.h>
#include <hip/hip_bf16.h>
#include <math.h>

// ---------------------------------------------------------------------------
// cosFormer linear attention on gfx1250: bf16 WMMA + async global->LDS staging.
// B=8, T=12, N=4096, D=64, dm=768, 2*dm=1536.
// All GEMM operands pre-converted to bf16 in contiguous row-major layouts so
// every LDS tile is staged with GLOBAL_LOAD_ASYNC_TO_LDS_B128 (ASYNCcnt),
// double-buffered, feeding v_wmma_f32_16x16x32_bf16.
// ---------------------------------------------------------------------------

typedef __attribute__((ext_vector_type(16))) __bf16 v16bf;
typedef __attribute__((ext_vector_type(8)))  float  v8f;

#define BM 128
#define BN 128
#define BK 64

#define NB    8
#define NT    12
#define NN    4096
#define ND    64
#define DM    768
#define DM2   1536

// 16-byte async copy: global -> LDS (per-lane). VDST = LDS address VGPR,
// VADDR = 64-bit global address, "off" = no SADDR. Tracked by ASYNCcnt.
__device__ __forceinline__ void async_copy16(unsigned lds_addr, const __bf16* gsrc) {
  asm volatile("global_load_async_to_lds_b128 %0, %1, off"
               :: "v"(lds_addr), "v"((unsigned long long)(size_t)gsrc)
               : "memory");
}

// ---------------------------------------------------------------------------
// Prologue: x [B,T,N,D] f32 -> x2bf [B,N,768] bf16 (fused transpose+convert)
// ---------------------------------------------------------------------------
__global__ __launch_bounds__(256) void cvt_x_kernel(const float* __restrict__ x,
                                                    __bf16* __restrict__ x2bf) {
  const size_t i = (size_t)blockIdx.x * 256 + threadIdx.x;  // over B*N*768
  const int d = (int)(i % DM);
  const size_t bn = i / DM;
  const int n = (int)(bn % NN);
  const int b = (int)(bn / NN);
  x2bf[i] = (__bf16)x[(((size_t)(b * NT + (d >> 6)) * NN) + n) * ND + (d & 63)];
}

// Prologue: W [768,768] f32 (row-major [out,in]) -> Wt [768,768] bf16 [in,out]
__global__ __launch_bounds__(256) void cvt_w_kernel(const float* __restrict__ W,
                                                    __bf16* __restrict__ Wt) {
  const int i = blockIdx.x * 256 + threadIdx.x;   // over 768*768, i = d*768 + j
  const int j = i % DM;
  const int d = i / DM;
  Wt[i] = (__bf16)W[(size_t)j * DM + d];
}

// ---------------------------------------------------------------------------
// Generic tiled WMMA GEMM, all-bf16 operands, async double-buffered staging.
//  OP 0: q proj  -> qbuf [B,N,1536]  (bias, relu*x, sin/cos split)
//  OP 1: k proj  -> kT   [B,1536,N]  (bias, relu*x, sin/cos split, transposed)
//  OP 2: v proj  -> vbuf [B,N,768]   (bias)
//  OP 3: kv = k_^T @ v (M=1536,N=768,K=4096) -> kvbuf [B,1536,768]
//  OP 4: attn = q_ @ kv (K=1536) ; *z ; +res -> arbuf [B,N,768]
//  OP 5: out = attnres @ WoT + bo ; scatter [B,T,N,D] f32 -> d_out
// Block 256 threads = 8 waves; block tile 128x128x64; wave tile 32x64.
// ---------------------------------------------------------------------------
template <int OP>
__global__ __launch_bounds__(256) void gemm_op(
    const __bf16* __restrict__ Abase, size_t astride, int lda,
    const __bf16* __restrict__ Bbase, size_t bstride, int ldb,
    const float*  __restrict__ xres,   // residual source (OP4)
    const float*  __restrict__ zbuf,   // 1/denominator (OP4)
    const float*  __restrict__ bias,   // OP0-2, OP5
    __bf16*       __restrict__ obf,    // bf16 output (OP0..4)
    float*        __restrict__ ofp,    // f32 output (OP5)
    int K)
{
  // A: WMMA fragment order [buf(2)][kstep(2)][mt(8)][lane(32)][elem(16)]
  __shared__ alignas(16) __bf16 As[2 * 2 * 8 * 32 * 16];   // 32 KB
  // B: row-major [buf(2)][kstep(2)][k'(32)][n(128)]
  __shared__ alignas(16) __bf16 Bs[2 * 2 * 32 * 128];      // 32 KB

  const int tid  = threadIdx.x;
  const int lane = tid & 31;
  const int wave = tid >> 5;
  const int wm   = wave >> 1;            // 0..3
  const int wn   = wave & 1;             // 0..1
  const int bn0  = blockIdx.x * BN;
  const int bm0  = blockIdx.y * BM;
  const int b    = blockIdx.z;

  const __bf16* Apt = Abase + (size_t)b * astride;
  const __bf16* Bpt = Bbase + (size_t)b * bstride;

  v8f acc[2][4];
#pragma unroll
  for (int i = 0; i < 2; i++)
#pragma unroll
    for (int j = 0; j < 4; j++)
#pragma unroll
      for (int e = 0; e < 8; e++) acc[i][j][e] = 0.0f;

  // Issue one K-tile of async copies: 4 A-chunks + 4 B-chunks of 16B per thread.
  auto stage = [&](int buf, int k0) {
#pragma unroll
    for (int cc = 0; cc < 4; cc++) {
      // ---- A: 128 rows x 64 k = 1024 chunks of 8 bf16 ----
      {
        const int c   = tid + cc * 256;       // 0..1023
        const int row = c & 127;
        const int kc8 = c >> 7;               // 0..7 (8-wide k chunk)
        const __bf16* g = Apt + (size_t)(bm0 + row) * lda + (k0 + kc8 * 8);
        const int mt    = row >> 4;
        const int dlane = (row & 15) + (kc8 & 1) * 16;
        const int ebase = ((kc8 >> 1) & 1) * 8;
        const int ks    = kc8 >> 2;
        const unsigned la = (unsigned)(size_t)
            &As[(((buf * 2 + ks) * 8 + mt) * 32 + dlane) * 16 + ebase];
        async_copy16(la, g);
      }
      // ---- B: 64 k x 128 n = 1024 chunks of 8 bf16 ----
      {
        const int c  = tid + cc * 256;
        const int n8 = c & 15;                // 0..15
        const int kk = c >> 4;                // 0..63
        const __bf16* g = Bpt + (size_t)(k0 + kk) * ldb + (bn0 + n8 * 8);
        const unsigned lb = (unsigned)(size_t)
            &Bs[((buf * 2 + (kk >> 5)) * 32 + (kk & 31)) * 128 + n8 * 8];
        async_copy16(lb, g);
      }
    }
  };

  const int ntiles = K >> 6;                 // K multiple of 64 for all OPs
  stage(0, 0);
  for (int t = 0; t < ntiles; t++) {
    if (t + 1 < ntiles) {
      stage((t + 1) & 1, (t + 1) << 6);
      asm volatile("s_wait_asynccnt 0x8" ::: "memory");  // tile t complete
    } else {
      asm volatile("s_wait_asynccnt 0x0" ::: "memory");
    }
    __syncthreads();

    const int buf = t & 1;
#pragma unroll
    for (int ks = 0; ks < 2; ks++) {
      v16bf af[2], bfr[4];
#pragma unroll
      for (int i = 0; i < 2; i++)
        af[i] = *reinterpret_cast<const v16bf*>(
            &As[(((buf * 2 + ks) * 8 + (wm * 2 + i)) * 32 + lane) * 16]);
#pragma unroll
      for (int j = 0; j < 4; j++)
#pragma unroll
        for (int s = 0; s < 16; s++)
          bfr[j][s] = Bs[((buf * 2 + ks) * 32 + ((s & 8) << 1) + ((lane >> 4) << 3) + (s & 7)) * 128 + (wn * 4 + j) * 16 + (lane & 15)];
#pragma unroll
      for (int i = 0; i < 2; i++)
#pragma unroll
        for (int j = 0; j < 4; j++)
          acc[i][j] = __builtin_amdgcn_wmma_f32_16x16x32_bf16(
              false, af[i], false, bfr[j], (short)0, acc[i][j], false, false);
    }
    __syncthreads();   // all waves done reading buf before it is re-staged
  }

  // ---- epilogue: lane l, vgpr v -> row=(l>>4)*8+v, col=l&15 per 16x16 tile ----
#pragma unroll
  for (int i = 0; i < 2; i++) {
#pragma unroll
    for (int j = 0; j < 4; j++) {
#pragma unroll
      for (int v = 0; v < 8; v++) {
        const int r = bm0 + wm * 32 + i * 16 + ((lane >> 4) << 3) + v;
        const int c = bn0 + wn * 64 + j * 16 + (lane & 15);
        float val = acc[i][j][v];
        if (OP <= 2 || OP == 5) val += bias[c];
        if (OP == 0) {
          val = fmaxf(val, 0.0f) * val;                     // relu(x)*x
          const float ang = 1.57079632679f * (float)(r + 1) * (1.0f / (float)NN);
          const size_t base = ((size_t)b * NN + r) * DM2;
          obf[base + c]      = (__bf16)(val * __sinf(ang));
          obf[base + DM + c] = (__bf16)(val * __cosf(ang));
        } else if (OP == 1) {
          val = fmaxf(val, 0.0f) * val;
          const float ang = 1.57079632679f * (float)(r + 1) * (1.0f / (float)NN);
          const size_t base = (size_t)b * DM2 * NN + (size_t)r;  // r = n index
          obf[base + (size_t)c * NN]        = (__bf16)(val * __sinf(ang));
          obf[base + (size_t)(DM + c) * NN] = (__bf16)(val * __cosf(ang));
        } else if (OP == 2) {
          obf[((size_t)b * NN + r) * DM + c] = (__bf16)val;
        } else if (OP == 3) {
          obf[((size_t)b * DM2 + r) * DM + c] = (__bf16)val;
        } else if (OP == 4) {
          const float res = xres[(((size_t)(b * NT + (c >> 6)) * NN) + r) * ND + (c & 63)];
          const float zz  = zbuf[(size_t)b * NN + r];
          obf[((size_t)b * NN + r) * DM + c] = (__bf16)(val * zz + res);
        } else { // OP == 5: scatter to [B,T,N,D] f32
          ofp[(((size_t)(b * NT + (c >> 6)) * NN) + r) * ND + (c & 63)] = val;
        }
      }
    }
  }
}

// ksum[b,d] = sum_n kT[b,d,n] — one wave32 per contiguous row of 4096
__global__ __launch_bounds__(256) void ksum_kernel(const __bf16* __restrict__ kT,
                                                   float* __restrict__ ksum) {
  const int rowid = blockIdx.x * 8 + (threadIdx.x >> 5);  // 0 .. 8*1536-1
  const int lane  = threadIdx.x & 31;
  const __bf16* p = kT + (size_t)rowid * NN;
  float s = 0.0f;
  for (int n = lane; n < NN; n += 32) s += (float)p[n];
#pragma unroll
  for (int off = 16; off; off >>= 1) s += __shfl_xor(s, off, 32);
  if (lane == 0) ksum[rowid] = s;
}

// z[b,n] = 1 / max(q_[b,n,:] . ksum[b,:], 1e-6) — one wave32 per row
__global__ __launch_bounds__(256) void z_kernel(const __bf16* __restrict__ qbuf,
                                                const float* __restrict__ ksum,
                                                float* __restrict__ zout) {
  const int row  = blockIdx.x * 8 + (threadIdx.x >> 5);   // 0 .. 8*4096-1
  const int lane = threadIdx.x & 31;
  const int b    = row >> 12;
  const __bf16* q  = qbuf + (size_t)row * DM2;
  const float*  ks = ksum + (size_t)b * DM2;
  float s = 0.0f;
  for (int d = lane; d < DM2; d += 32) s += (float)q[d] * ks[d];
#pragma unroll
  for (int off = 16; off; off >>= 1) s += __shfl_xor(s, off, 32);
  if (lane == 0) zout[row] = 1.0f / fmaxf(s, 1e-6f);
}

extern "C" void kernel_launch(void* const* d_in, const int* in_sizes, int n_in,
                              void* d_out, int out_size, void* d_ws, size_t ws_size,
                              hipStream_t stream) {
  const float* x  = (const float*)d_in[0];
  const float* Wq = (const float*)d_in[1];
  const float* bq = (const float*)d_in[2];
  const float* Wk = (const float*)d_in[3];
  const float* bk = (const float*)d_in[4];
  const float* Wv = (const float*)d_in[5];
  const float* bv = (const float*)d_in[6];
  const float* Wo = (const float*)d_in[7];
  const float* bo = (const float*)d_in[8];
  float* out = (float*)d_out;

  // ---- workspace layout ----
  char* ws = (char*)d_ws;
  const size_t SZ_X2 = (size_t)NB * NN * DM  * sizeof(__bf16);  //  50331648
  const size_t SZ_W  = (size_t)DM * DM       * sizeof(__bf16);  //   1179648
  const size_t SZ_QK = (size_t)NB * NN * DM2 * sizeof(__bf16);  // 100663296
  const size_t SZ_V  = (size_t)NB * NN * DM  * sizeof(__bf16);  //  50331648
  const size_t SZ_KV = (size_t)NB * DM2 * DM * sizeof(__bf16);  //  18874368
  size_t off = 0;
  __bf16* x2bf  = (__bf16*)(ws + off); off += SZ_X2;
  __bf16* wtq   = (__bf16*)(ws + off); off += SZ_W;
  __bf16* wtk   = (__bf16*)(ws + off); off += SZ_W;
  __bf16* wtv   = (__bf16*)(ws + off); off += SZ_W;
  __bf16* wto   = (__bf16*)(ws + off); off += SZ_W;
  __bf16* qbuf  = (__bf16*)(ws + off); off += SZ_QK;   // [B,N,1536]
  __bf16* kT    = (__bf16*)(ws + off); off += SZ_QK;   // [B,1536,N]
  __bf16* vbuf  = (__bf16*)(ws + off); off += SZ_V;    // [B,N,768]
  __bf16* kvbuf = (__bf16*)(ws + off); off += SZ_KV;   // [B,1536,768]
  __bf16* arbuf = (__bf16*)(ws + off); off += SZ_V;    // [B,N,768]
  float*  ksum  = (float*) (ws + off); off += (size_t)NB * DM2 * sizeof(float);
  float*  zbuf  = (float*) (ws + off);

  const dim3 blk(256);
  // prologue conversions
  cvt_x_kernel<<<(NB * NN * DM) / 256, blk, 0, stream>>>(x, x2bf);
  cvt_w_kernel<<<(DM * DM) / 256, blk, 0, stream>>>(Wq, wtq);
  cvt_w_kernel<<<(DM * DM) / 256, blk, 0, stream>>>(Wk, wtk);
  cvt_w_kernel<<<(DM * DM) / 256, blk, 0, stream>>>(Wv, wtv);
  cvt_w_kernel<<<(DM * DM) / 256, blk, 0, stream>>>(Wo, wto);

  // q / k / v projections: M=4096, N=768, K=768
  gemm_op<0><<<dim3(DM / BN, NN / BM, NB), blk, 0, stream>>>(
      x2bf, (size_t)NN * DM, DM, wtq, 0, DM,
      nullptr, nullptr, bq, qbuf, nullptr, DM);
  gemm_op<1><<<dim3(DM / BN, NN / BM, NB), blk, 0, stream>>>(
      x2bf, (size_t)NN * DM, DM, wtk, 0, DM,
      nullptr, nullptr, bk, kT, nullptr, DM);
  gemm_op<2><<<dim3(DM / BN, NN / BM, NB), blk, 0, stream>>>(
      x2bf, (size_t)NN * DM, DM, wtv, 0, DM,
      nullptr, nullptr, bv, vbuf, nullptr, DM);
  // ksum + z
  ksum_kernel<<<(NB * DM2) / 8, blk, 0, stream>>>(kT, ksum);
  z_kernel<<<(NB * NN) / 8, blk, 0, stream>>>(qbuf, ksum, zbuf);
  // kv = k_^T @ v : M=1536, N=768, K=4096
  gemm_op<3><<<dim3(DM / BN, DM2 / BM, NB), blk, 0, stream>>>(
      kT, (size_t)DM2 * NN, NN, vbuf, (size_t)NN * DM, DM,
      nullptr, nullptr, nullptr, kvbuf, nullptr, NN);
  // attn = q_ @ kv, *z, +res : M=4096, N=768, K=1536
  gemm_op<4><<<dim3(DM / BN, NN / BM, NB), blk, 0, stream>>>(
      qbuf, (size_t)NN * DM2, DM2, kvbuf, (size_t)DM2 * DM, DM,
      x, zbuf, nullptr, arbuf, nullptr, DM2);
  // out = attnres @ Wo^T + bo, scatter to [B,T,N,D] : M=4096, N=768, K=768
  gemm_op<5><<<dim3(DM / BN, NN / BM, NB), blk, 0, stream>>>(
      arbuf, (size_t)NN * DM, DM, wto, 0, DM,
      nullptr, nullptr, bo, nullptr, out, DM);
}
